// SASRecIHM_40888088658109
// MI455X (gfx1250) — compile-verified, matrix operand order checked by reading, not run on hardware
//
#include <hip/hip_runtime.h>

// ---------------- MI455X / gfx1250 SASRec-IHM forward ----------------
// All GEMMs run on v_wmma_f32_16x16x32_f16 (f16 in, f32 accum).
// Streaming GEMM input tiles are staged to LDS via the Tensor Data Mover
// (tensor_load_to_lds + s_wait_tensorcnt) when the builtin is available.

typedef __attribute__((ext_vector_type(16))) _Float16     v16h;
typedef __attribute__((ext_vector_type(8)))  float        v8f;
typedef __attribute__((ext_vector_type(4)))  unsigned int u32x4_t;
typedef __attribute__((ext_vector_type(8)))  int          i32x8_t;
typedef __attribute__((ext_vector_type(4)))  int          i32x4_t;

#define B_     64
#define L_     200
#define M_     32
#define H_     128
#define E_     8
#define NH_    2
#define NTOK   12800        // B*L
#define ENROWS 102400       // E*NTOK
#define SQRT_H 11.313708498984760f

#if defined(__HIP_DEVICE_COMPILE__) && __has_builtin(__builtin_amdgcn_tensor_load_to_lds) && __has_builtin(__builtin_amdgcn_s_wait_tensorcnt)
#define USE_TDM 1
#else
#define USE_TDM 0
#endif

// -------- WMMA fragment load (16-bit A layout, ISA 7.12.2) --------
// lane&15 selects the row (A: M row; B: N column from a K-transposed tile),
// lane>>4 selects the 8-wide K half; VGPR v holds K pair {2v | 16+2(v-4)}.
__device__ __forceinline__ v16h wmma_load_frag(const _Float16* tile, int ld,
                                               int row, int kbase, int hsel) {
  const _Float16* p = tile + row * ld + kbase + 8 * hsel;
  v16h f;
#pragma unroll
  for (int v = 0; v < 4; ++v) { f[2*v] = p[2*v]; f[2*v+1] = p[2*v+1]; }
#pragma unroll
  for (int v = 0; v < 4; ++v) { f[8+2*v] = p[16+2*v]; f[9+2*v] = p[17+2*v]; }
  return f;
}

#define WMMA_F16(A, B, C) \
  __builtin_amdgcn_wmma_f32_16x16x32_f16(false, (A), false, (B), (short)0, (C), false, false)

// -------- TDM: 1-D linear 16384-float (64KB) global -> LDS copy --------
#if USE_TDM
__device__ __forceinline__ void tdm_load_16k_f32(const float* gsrc, float* ldst) {
  unsigned long long ga = (unsigned long long)(size_t)gsrc;
  unsigned lds = (unsigned)(size_t)(void*)ldst;   // LDS aperture: low 32 bits = LDS offset
  u32x4_t g0;
  g0[0] = 1u;                                          // count=1 (valid descriptor)
  g0[1] = lds;                                         // lds_addr
  g0[2] = (unsigned)(ga & 0xffffffffull);              // global_addr[31:0]
  g0[3] = (unsigned)((ga >> 32) & 0x1ffffffull) | (2u << 30);  // addr[56:32] | type=2
  i32x8_t g1;
  g1[0] = (int)(2u << 16);                             // data_size = 4B
  g1[1] = (int)(16384u << 16);                         // tensor_dim0 lo16
  g1[2] = (int)((16384u >> 16) | (1u << 16));          // dim0 hi16 | tensor_dim1 = 1
  g1[3] = (int)(16384u << 16);                         // dim1 hi16 | tile_dim0 = 16384
  g1[4] = 1;                                           // tile_dim1 = 1, tile_dim2 = 0
  g1[5] = 16384;                                       // tensor_dim0_stride lo32
  g1[6] = 0;
  g1[7] = 0;
  i32x4_t z4 = {0, 0, 0, 0};
  i32x8_t z8 = {0, 0, 0, 0, 0, 0, 0, 0};
  // 6-arg form (clang-23 / therock-10 headers): groups 2/3 + extra group unused.
  __builtin_amdgcn_tensor_load_to_lds(g0, g1, z4, z4, z8, 0);
  __builtin_amdgcn_s_wait_tensorcnt(0);
}
#endif

// -------- 128-thread block reduction (4 wave32 waves) --------
__device__ __forceinline__ float block_reduce_128(float v, volatile float* red) {
#pragma unroll
  for (int mm = 16; mm >= 1; mm >>= 1) v += __shfl_xor(v, mm, 32);
  const int w = threadIdx.x >> 5;
  if ((threadIdx.x & 31) == 0) red[w] = v;
  __syncthreads();
  float r = red[0] + red[1] + red[2] + red[3];
  __syncthreads();
  return r;
}

// =====================================================================
// embed: seqs = keep * (item[log]*sqrt(H) + pos_table)
// =====================================================================
__global__ __launch_bounds__(128) void embed_kernel(
    const int* __restrict__ logs, const float* __restrict__ item_tab,
    const float* __restrict__ pos_tab, float* __restrict__ seqs,
    float* __restrict__ keep) {
  const int n = blockIdx.x, t = threadIdx.x;
  const int idx = logs[n];
  const int l = n % L_;
  float v = 0.f;
  if (idx != 0) v = item_tab[(size_t)idx * H_ + t] * SQRT_H + pos_tab[l * H_ + t];
  seqs[(size_t)n * H_ + t] = v;
  if (t == 0) keep[n] = (idx != 0) ? 1.f : 0.f;
}

// =====================================================================
// LayerNorm over H=128 (eps=1e-8), one row per block
// =====================================================================
__global__ __launch_bounds__(128) void ln_kernel(
    const float* __restrict__ X, const float* __restrict__ s,
    const float* __restrict__ b, float* __restrict__ Y) {
  __shared__ float red[4];
  const size_t row = blockIdx.x;
  const int t = threadIdx.x;
  float x = X[row * H_ + t];
  float mean = block_reduce_128(x, red) * (1.0f / H_);
  float d = x - mean;
  float var = block_reduce_128(d * d, red) * (1.0f / H_);
  Y[row * H_ + t] = d * rsqrtf(var + 1e-8f) * s[t] + b[t];
}

// =====================================================================
// Generic WMMA GEMM: Y[N,128] = f(X[N,128] @ W[128,128] + bias) [+res][*rowmask]
// One workgroup = 128x128 output tile, 8 waves x (16 rows x 128 cols).
// X tile staged to LDS by the TDM, W staged K-transposed as f16.
// =====================================================================
__global__ __launch_bounds__(256) void gemm128_kernel(
    const float* __restrict__ X, const float* __restrict__ W,
    const float* __restrict__ bias, const float* __restrict__ residual,
    const float* __restrict__ rowmask, float* __restrict__ Y, int do_relu) {
  extern __shared__ char smem[];
  float*    xstage = (float*)smem;                   // [128][128] f32 (TDM dest)
  _Float16* a16    = (_Float16*)(smem + 16384 * 4);  // [128][136] f16
  _Float16* bt16   = a16 + 128 * 136;                // [128][136] f16 (W transposed)

  const size_t rowbase = (size_t)blockIdx.x * 128;
  const float* Xtile = X + rowbase * 128;
  __builtin_prefetch(W, 0, 1);   // global_prefetch_b8 of the weight tile

#if USE_TDM
  if (threadIdx.x < 32) tdm_load_16k_f32(Xtile, xstage);
#else
  for (int i = threadIdx.x; i < 16384; i += 256) xstage[i] = Xtile[i];
#endif
  for (int i = threadIdx.x; i < 16384; i += 256) {  // W[k][n] -> bt16[n][k]
    int k = i >> 7, n = i & 127;
    bt16[n * 136 + k] = (_Float16)W[i];
  }
  __syncthreads();
  for (int i = threadIdx.x; i < 16384; i += 256) {
    int r = i >> 7, c = i & 127;
    a16[r * 136 + c] = (_Float16)xstage[i];
  }
  __syncthreads();

  const int lane = threadIdx.x & 31;
  const int wv = threadIdx.x >> 5;
  const int nsel = lane & 15, hsel = lane >> 4;

  v8f acc[8] = {};
#pragma unroll
  for (int kstep = 0; kstep < 4; ++kstep) {
    v16h af = wmma_load_frag(a16, 136, wv * 16 + nsel, kstep * 32, hsel);
#pragma unroll
    for (int ct = 0; ct < 8; ++ct) {
      v16h bf = wmma_load_frag(bt16, 136, ct * 16 + nsel, kstep * 32, hsel);
      acc[ct] = WMMA_F16(af, bf, acc[ct]);
    }
  }
#pragma unroll
  for (int ct = 0; ct < 8; ++ct) {
    const int col = ct * 16 + nsel;
    const float bc = bias ? bias[col] : 0.0f;
#pragma unroll
    for (int r = 0; r < 8; ++r) {
      const size_t grow = rowbase + wv * 16 + hsel * 8 + r;
      float v = acc[ct][r] + bc;
      if (residual) v += residual[grow * 128 + col];
      if (do_relu)  v = fmaxf(v, 0.0f);
      if (rowmask)  v *= rowmask[grow];
      Y[grow * 128 + col] = v;
    }
  }
}

// =====================================================================
// Causal flash attention, one workgroup per (batch, head). L=200, hd=64.
// Scores and P@V both via WMMA; online softmax with 16-lane shfl reductions.
// =====================================================================
__global__ __launch_bounds__(256) void attn_kernel(
    const float* __restrict__ Q, const float* __restrict__ K,
    const float* __restrict__ V, float* __restrict__ O) {
  extern __shared__ char smem[];
  _Float16* qs  = (_Float16*)smem;       // [224][72]  (q * 1/sqrt(hd))
  _Float16* ksm = qs + 224 * 72;         // [224][72]
  _Float16* vt  = ksm + 224 * 72;        // [64][224]  (V transposed)
  _Float16* ps  = vt + 64 * 224;         // [8][16*32] per-wave P scratch

  const int b = blockIdx.x >> 1;
  const int hh = blockIdx.x & 1;
  const int lane = threadIdx.x & 31;
  const int wv = threadIdx.x >> 5;
  const int nsel = lane & 15, hsel = lane >> 4;
  _Float16* ph = ps + wv * 512;

  for (int i = threadIdx.x; i < 224 * 64; i += 256) {
    int r = i >> 6, d = i & 63;
    float qv = 0.f, kv = 0.f, vv = 0.f;
    if (r < L_) {
      size_t g = ((size_t)(b * L_ + r)) * H_ + hh * 64 + d;
      qv = Q[g] * 0.125f;  kv = K[g];  vv = V[g];
    }
    qs[r * 72 + d]  = (_Float16)qv;
    ksm[r * 72 + d] = (_Float16)kv;
    vt[d * 224 + r] = (_Float16)vv;
  }
  __syncthreads();

  for (int rt = wv; rt < 13; rt += 8) {       // 13 row tiles of 16
    v8f o0 = {}, o1 = {}, o2 = {}, o3 = {};
    float mrow[8], lrow[8];
#pragma unroll
    for (int r = 0; r < 8; ++r) { mrow[r] = -1e30f; lrow[r] = 0.f; }
    const int growb = rt * 16 + hsel * 8;

    for (int ct0 = 0; ct0 <= rt; ct0 += 2) {  // column tiles in pairs
      v8f s0 = {}, s1 = {};
#pragma unroll
      for (int k2 = 0; k2 < 2; ++k2) {        // hd=64 -> 2 K-steps of 32
        v16h aq  = wmma_load_frag(qs, 72, rt * 16 + nsel, k2 * 32, hsel);
        v16h bk0 = wmma_load_frag(ksm, 72, ct0 * 16 + nsel, k2 * 32, hsel);
        s0 = WMMA_F16(aq, bk0, s0);
        v16h bk1 = wmma_load_frag(ksm, 72, (ct0 + 1) * 16 + nsel, k2 * 32, hsel);
        s1 = WMMA_F16(aq, bk1, s1);
      }
      const int c0 = ct0 * 16 + nsel, c1 = c0 + 16;
#pragma unroll
      for (int r = 0; r < 8; ++r) {
        const int grow = growb + r;
        float v0 = (c0 <= grow) ? s0[r] : -1e30f;   // causal mask
        float v1 = (c1 <= grow) ? s1[r] : -1e30f;
        float t = fmaxf(v0, v1);
#pragma unroll
        for (int mm = 8; mm >= 1; mm >>= 1) t = fmaxf(t, __shfl_xor(t, mm, 32));
        const float mn = fmaxf(mrow[r], t);
        const float sc = __expf(mrow[r] - mn);
        const float p0 = __expf(v0 - mn);
        const float p1 = __expf(v1 - mn);
        float psum = p0 + p1;
#pragma unroll
        for (int mm = 8; mm >= 1; mm >>= 1) psum += __shfl_xor(psum, mm, 32);
        lrow[r] = lrow[r] * sc + psum;
        mrow[r] = mn;
        o0[r] *= sc; o1[r] *= sc; o2[r] *= sc; o3[r] *= sc;
        ph[(hsel * 8 + r) * 32 + nsel]      = (_Float16)p0;   // C->A relayout via LDS
        ph[(hsel * 8 + r) * 32 + 16 + nsel] = (_Float16)p1;
      }
      v16h pa = wmma_load_frag(ph, 32, nsel, 0, hsel);
      v16h bv0 = wmma_load_frag(vt, 224, 0 * 16 + nsel, ct0 * 16, hsel);
      o0 = WMMA_F16(pa, bv0, o0);
      v16h bv1 = wmma_load_frag(vt, 224, 1 * 16 + nsel, ct0 * 16, hsel);
      o1 = WMMA_F16(pa, bv1, o1);
      v16h bv2 = wmma_load_frag(vt, 224, 2 * 16 + nsel, ct0 * 16, hsel);
      o2 = WMMA_F16(pa, bv2, o2);
      v16h bv3 = wmma_load_frag(vt, 224, 3 * 16 + nsel, ct0 * 16, hsel);
      o3 = WMMA_F16(pa, bv3, o3);
    }
#pragma unroll
    for (int r = 0; r < 8; ++r) {
      const int grow = growb + r;
      if (grow < L_) {
        const float inv = 1.0f / lrow[r];
        const size_t base = ((size_t)(b * L_ + grow)) * H_ + hh * 64;
        O[base + nsel]      = o0[r] * inv;
        O[base + 16 + nsel] = o1[r] * inv;
        O[base + 32 + nsel] = o2[r] * inv;
        O[base + 48 + nsel] = o3[r] * inv;
      }
    }
  }
}

// =====================================================================
// IHM: per workgroup, 4 tokens n. Gather V (33 rows, L2-resident tables),
// k = V@Wk+bk via WMMA (144x128x128), last-row q, masked softmax over 33,
// weighted sum -> ihm_out[E][N][H]. Raw item embedding saved for gate/dot.
// =====================================================================
__global__ __launch_bounds__(256) void ihm_kernel(
    const int* __restrict__ iseq, const int* __restrict__ ihseq,
    const float* __restrict__ item_tab, const float* __restrict__ user_tab,
    const float* __restrict__ Wq, const float* __restrict__ bq,
    const float* __restrict__ Wk, const float* __restrict__ bk,
    float* __restrict__ ihm_out, float* __restrict__ rawb) {
  extern __shared__ char smem[];
  float* kls = (float*)smem;              // [144][128] k projection
  float* qls = kls + 144 * 128;           // [4][128]
  float* sls = qls + 4 * 128;             // [4][8][36] scores/attn
  float* mls = sls + 4 * 8 * 36;          // [4][36] mask
  _Float16* vf = (_Float16*)(mls + 4 * 36);  // [144][136] V f16
  _Float16* wt = vf + 144 * 136;             // [128][136] Wk transposed f16

  const int n0 = blockIdx.x * 4;
  const int lane = threadIdx.x & 31;
  const int wv = threadIdx.x >> 5;
  const int nsel = lane & 15, hsel = lane >> 4;

  for (int i = threadIdx.x; i < H_ * H_; i += 256) {   // Wk[k][n] -> wt[n][k]
    int k = i >> 7, n = i & 127;
    wt[n * 136 + k] = (_Float16)Wk[i];
  }
  for (int i = threadIdx.x; i < 4 * 36 * H_; i += 256) {  // gather V
    const int d = i & 127;
    const int row = i >> 7;                 // 0..143
    const int g = row / 36, m = row % 36;
    const int n = n0 + g;
    float v = 0.f;
    if (m < M_) {
      const int idx = ihseq[(size_t)n * M_ + m];
      if (idx != 0) v = user_tab[(size_t)idx * H_ + d] * SQRT_H;
      if (d == 0) mls[g * 36 + m] = (idx == 0) ? 1.f : 0.f;
    } else if (m == M_) {
      const int idx = iseq[n];
      const float raw = (idx != 0) ? item_tab[(size_t)idx * H_ + d] : 0.f;
      rawb[(size_t)n * H_ + d] = raw;
      v = raw * (float)H_;                  // concat'ed item token ends up *H
      if (d == 0) mls[g * 36 + m] = 0.f;
    } else {
      if (d == 0) mls[g * 36 + m] = 1.f;    // pad rows
    }
    vf[row * 136 + d] = (_Float16)v;
  }
  __syncthreads();

  // k = V @ Wk + bk   (9 M-tiles x 8 N-tiles of 16x16, K=128)
  for (int job = wv; job < 72; job += 8) {
    const int mt = job >> 3, ct = job & 7;
    v8f acc = {};
#pragma unroll
    for (int kstep = 0; kstep < 4; ++kstep) {
      v16h af = wmma_load_frag(vf, 136, mt * 16 + nsel, kstep * 32, hsel);
      v16h bf = wmma_load_frag(wt, 136, ct * 16 + nsel, kstep * 32, hsel);
      acc = WMMA_F16(af, bf, acc);
    }
    const int col = ct * 16 + nsel;
    const float bc = bk[col];
#pragma unroll
    for (int r = 0; r < 8; ++r)
      kls[(mt * 16 + hsel * 8 + r) * 128 + col] = acc[r] + bc;
  }
  __syncthreads();

  // q = V[last] @ Wq + bq  (one 128-vector per token)
  for (int idx = threadIdx.x; idx < 4 * H_; idx += 256) {
    const int g = idx >> 7, hcol = idx & 127;
    float s = bq[hcol];
    const _Float16* vr = vf + (g * 36 + 32) * 136;
    for (int d = 0; d < H_; ++d) s += (float)vr[d] * Wq[d * H_ + hcol];
    qls[g * 128 + hcol] = s;
  }
  __syncthreads();

  // scores s[g][e][m] = (q_e . k_me)/4, masked
  for (int idx = threadIdx.x; idx < 4 * E_ * 33; idx += 256) {
    const int m = idx % 33;
    const int e = (idx / 33) & 7;
    const int g = idx / 264;
    float s = 0.f;
    const float* qp = qls + g * 128 + e * 16;
    const float* kp = kls + (g * 36 + m) * 128 + e * 16;
#pragma unroll
    for (int d = 0; d < 16; ++d) s += qp[d] * kp[d];
    s *= 0.25f;
    if (mls[g * 36 + m] != 0.f) s = -1e30f;
    sls[(g * 8 + e) * 36 + m] = s;
  }
  __syncthreads();

  if (threadIdx.x < 32) {                 // softmax over 33 per (g,e)
    float* row = sls + threadIdx.x * 36;
    float mx = -1e30f;
    for (int m = 0; m < 33; ++m) mx = fmaxf(mx, row[m]);
    float den = 0.f;
    for (int m = 0; m < 33; ++m) { float p = __expf(row[m] - mx); row[m] = p; den += p; }
    const float inv = 1.f / den;
    for (int m = 0; m < 33; ++m) row[m] *= inv;
  }
  __syncthreads();

  // ihm[e][n][h] = sum_m a[g][e][m] * V[g][m][h]
  for (int idx = threadIdx.x; idx < 4 * E_ * H_; idx += 256) {
    const int hcol = idx & 127;
    const int e = (idx >> 7) & 7;
    const int g = idx >> 10;
    const float* arow = sls + (g * 8 + e) * 36;
    const _Float16* vcol = vf + g * 36 * 136 + hcol;
    float acc = 0.f;
    for (int m = 0; m < 33; ++m) acc += arow[m] * (float)vcol[m * 136];
    ihm_out[((size_t)e * NTOK + (n0 + g)) * H_ + hcol] = acc;
  }
}

// =====================================================================
// Gate over experts + final dot products with feats.
// =====================================================================
__global__ __launch_bounds__(128) void gate_kernel(
    const float* __restrict__ feats, const float* __restrict__ ihm,
    const float* __restrict__ rawb, const float* __restrict__ gw,
    const float* __restrict__ gb, float* __restrict__ outMain,
    float* __restrict__ outRaw) {
  __shared__ float red[4];
  __shared__ float ge[8];
  const int n = blockIdx.x, t = threadIdx.x;
  const float fv = feats[(size_t)n * H_ + t];
  const float rv = rawb[(size_t)n * H_ + t];
  const float item_e = rv * SQRT_H;
  const float p1 = item_e * gw[t];
  for (int e = 0; e < E_; ++e) {
    float x = p1 + ihm[((size_t)e * NTOK + n) * H_ + t] * gw[H_ + t];
    float s = block_reduce_128(x, red);
    if (t == 0) ge[e] = s + gb[0];
  }
  __syncthreads();
  float mx = ge[0];
#pragma unroll
  for (int e = 1; e < E_; ++e) mx = fmaxf(mx, ge[e]);
  float den = 0.f, wgt[E_];
#pragma unroll
  for (int e = 0; e < E_; ++e) { wgt[e] = __expf(ge[e] - mx); den += wgt[e]; }
  const float inv = 1.f / den;
  float pe = 0.f;
#pragma unroll
  for (int e = 0; e < E_; ++e)
    pe += wgt[e] * inv * ihm[((size_t)e * NTOK + n) * H_ + t];
  const float o1 = block_reduce_128(fv * pe, red);
  const float o2 = block_reduce_128(fv * rv, red);
  if (t == 0) { outMain[n] = o1; outRaw[n] = o2; }
}

// =====================================================================
#define GEMM_SMEM (16384 * 4 + 2 * 128 * 136 * 2)                  // 135168 B
#define ATT_SMEM  (2 * 224 * 72 * 2 + 64 * 224 * 2 + 8 * 512 * 2)  // 101376 B
#define IHM_SMEM  ((144 * 128 + 4 * 128 + 4 * 8 * 36 + 4 * 36) * 4 + \
                   (144 * 136 + 128 * 136) * 2)                    // 154944 B

extern "C" void kernel_launch(void* const* d_in, const int* in_sizes, int n_in,
                              void* d_out, int out_size, void* d_ws, size_t ws_size,
                              hipStream_t stream) {
  (void)in_sizes; (void)n_in; (void)out_size; (void)ws_size;
  const int* log_seqs = (const int*)d_in[0];
  const int* pos_seqs = (const int*)d_in[1];
  const int* neg_seqs = (const int*)d_in[2];
  const int* pos_ih   = (const int*)d_in[3];
  const int* neg_ih   = (const int*)d_in[4];
  const float* item_table = (const float*)d_in[5];
  const float* user_table = (const float*)d_in[6];
  const float* pos_table  = (const float*)d_in[7];
  const float* ln1_s = (const float*)d_in[8];
  const float* ln1_b = (const float*)d_in[9];
  const float* Wq = (const float*)d_in[10];
  const float* Wk = (const float*)d_in[11];
  const float* Wv = (const float*)d_in[12];
  const float* bq = (const float*)d_in[13];
  const float* bk = (const float*)d_in[14];
  const float* bv = (const float*)d_in[15];
  const float* Wo = (const float*)d_in[16];
  const float* bo = (const float*)d_in[17];
  const float* ln2_s = (const float*)d_in[18];
  const float* ln2_b = (const float*)d_in[19];
  const float* c1w = (const float*)d_in[20];
  const float* c1b = (const float*)d_in[21];
  const float* c2w = (const float*)d_in[22];
  const float* c2b = (const float*)d_in[23];
  const float* last_s = (const float*)d_in[24];
  const float* last_b = (const float*)d_in[25];
  const float* iWq = (const float*)d_in[26];
  const float* ibq = (const float*)d_in[27];
  const float* iWk = (const float*)d_in[28];
  const float* ibk = (const float*)d_in[29];
  const float* iln_s = (const float*)d_in[30];
  const float* iln_b = (const float*)d_in[31];
  const float* ic1w = (const float*)d_in[32];
  const float* ic1b = (const float*)d_in[33];
  const float* ic2w = (const float*)d_in[34];
  const float* ic2b = (const float*)d_in[35];
  const float* gate_w = (const float*)d_in[36];
  const float* gate_b = (const float*)d_in[37];

  float* cur = (float*)d_ws;
  auto alloc = [&](size_t nf) { float* p = cur; cur += nf; return p; };
  const size_t NHf = (size_t)NTOK * H_;
  float* seqs  = alloc(NHf);
  float* keep  = alloc(NTOK);
  float* qin   = alloc(NHf);
  float* qb    = alloc(NHf);
  float* kbuf  = alloc(NHf);
  float* vbuf  = alloc(NHf);
  float* obuf  = alloc(NHf);
  float* t1    = alloc(NHf);
  float* h1    = alloc(NHf);
  float* feats = alloc(NHf);
  float* rawb  = alloc(NHf);
  float* ihmA  = alloc((size_t)ENROWS * H_);
  float* ihmB  = alloc((size_t)ENROWS * H_);
  float* ihmC  = alloc((size_t)ENROWS * H_);
  float* out = (float*)d_out;

  embed_kernel<<<NTOK, 128, 0, stream>>>(log_seqs, item_table, pos_table, seqs, keep);

  for (int i = 0; i < 2; ++i) {
    ln_kernel<<<NTOK, 128, 0, stream>>>(seqs, ln1_s + i * H_, ln1_b + i * H_, qin);
    gemm128_kernel<<<100, 256, GEMM_SMEM, stream>>>(qin,  Wq + i*H_*H_, bq + i*H_, nullptr, nullptr, qb, 0);
    gemm128_kernel<<<100, 256, GEMM_SMEM, stream>>>(seqs, Wk + i*H_*H_, bk + i*H_, nullptr, nullptr, kbuf, 0);
    gemm128_kernel<<<100, 256, GEMM_SMEM, stream>>>(seqs, Wv + i*H_*H_, bv + i*H_, nullptr, nullptr, vbuf, 0);
    attn_kernel<<<B_ * NH_, 256, ATT_SMEM, stream>>>(qb, kbuf, vbuf, obuf);
    gemm128_kernel<<<100, 256, GEMM_SMEM, stream>>>(obuf, Wo + i*H_*H_, bo + i*H_, qin, nullptr, seqs, 0);
    ln_kernel<<<NTOK, 128, 0, stream>>>(seqs, ln2_s + i * H_, ln2_b + i * H_, t1);
    gemm128_kernel<<<100, 256, GEMM_SMEM, stream>>>(t1, c1w + i*H_*H_, c1b + i*H_, nullptr, nullptr, h1, 1);
    gemm128_kernel<<<100, 256, GEMM_SMEM, stream>>>(h1, c2w + i*H_*H_, c2b + i*H_, t1, keep, seqs, 0);
  }
  ln_kernel<<<NTOK, 128, 0, stream>>>(seqs, last_s, last_b, feats);

  for (int mod = 0; mod < 2; ++mod) {
    const int* iseq = mod ? neg_seqs : pos_seqs;
    const int* ihs  = mod ? neg_ih : pos_ih;
    ihm_kernel<<<NTOK / 4, 256, IHM_SMEM, stream>>>(iseq, ihs, item_table, user_table,
                                                    iWq, ibq, iWk, ibk, ihmA, rawb);
    ln_kernel<<<ENROWS, 128, 0, stream>>>(ihmA, iln_s, iln_b, ihmB);
    gemm128_kernel<<<800, 256, GEMM_SMEM, stream>>>(ihmB, ic1w, ic1b, nullptr, nullptr, ihmC, 1);
    gemm128_kernel<<<800, 256, GEMM_SMEM, stream>>>(ihmC, ic2w, ic2b, ihmB, nullptr, ihmA, 0);
    gate_kernel<<<NTOK, 128, 0, stream>>>(feats, ihmA, rawb, gate_w, gate_b,
                                          out + mod * NTOK, out + (2 + mod) * NTOK);
  }
}
